// LinkPredictor_25400436589171
// MI455X (gfx1250) — matrix operand order repaired and checked by reading, not compiled, not run on hardware
//
#include <hip/hip_runtime.h>

#define DIM     128
#define NTILES  8      // DIM/16
#define KSTEPS  4      // DIM/32

typedef __attribute__((ext_vector_type(16))) __bf16 v16bf;
typedef __attribute__((ext_vector_type(8)))  __bf16 v8bf;
typedef __attribute__((ext_vector_type(8)))  float  v8f;

__device__ __forceinline__ unsigned short f2bf(float f) {
  unsigned int u = __float_as_uint(f);
  u += 0x7FFFu + ((u >> 16) & 1u);          // round-to-nearest-even
  return (unsigned short)(u >> 16);
}
__device__ __forceinline__ float bf2f(unsigned short h) {
  return __uint_as_float(((unsigned int)h) << 16);
}

// Native CDNA5 no-return f32 atomic add at device scope (RMW performed in L2
// atomic units; tracked via STOREcnt, drained by S_ENDPGM's implicit wait-idle).
__device__ __forceinline__ void fadd_dev(float* p, float v) {
  asm volatile("global_atomic_add_f32 %0, %1, off scope:SCOPE_DEV"
               :: "v"(p), "v"(v)
               : "memory");
}

// ---------------- degree / normalization ----------------
__global__ void k_deg_init(unsigned int* deg, int n) {
  int i = blockIdx.x * blockDim.x + threadIdx.x;
  if (i < n) deg[i] = 1u;                   // self loop
}
__global__ void k_deg_count(const int* __restrict__ dst, unsigned int* deg, int e) {
  int i = blockIdx.x * blockDim.x + threadIdx.x;
  if (i < e) atomicAdd(&deg[dst[i]], 1u);
}
__global__ void k_dinv(float* dinv_io, int n) {   // in-place uint -> float
  int i = blockIdx.x * blockDim.x + threadIdx.x;
  if (i < n) {
    unsigned int d = ((const unsigned int*)dinv_io)[i];
    dinv_io[i] = rsqrtf((float)d);
  }
}

// ---------------- bf16x3 split helpers ----------------
__global__ void k_cvt_split(const float* __restrict__ x,
                            unsigned short* __restrict__ hi,
                            unsigned short* __restrict__ lo, long long n) {
  long long i = (long long)blockIdx.x * blockDim.x + threadIdx.x;
  if (i < n) {
    float v = x[i];
    unsigned short h = f2bf(v);
    hi[i] = h;
    lo[i] = f2bf(v - bf2f(h));
  }
}

// Pack W (DIM x DIM row-major, W[k][n]) into per-lane B-fragment order:
// index = ((nt*KSTEPS + ks)*32 + lane)*16 + j ; col = nt*16+(lane&15),
// k = ks*32 + (lane>>4)*8 + (j&7) + (j>>3)*16   (matches 16-bit B 32x16 wave32 layout)
__global__ void k_pack_w(const float* __restrict__ W,
                         unsigned short* __restrict__ whi,
                         unsigned short* __restrict__ wlo) {
  int t = blockIdx.x * blockDim.x + threadIdx.x;
  if (t >= NTILES * KSTEPS * 32 * 16) return;
  int j    = t & 15;
  int lane = (t >> 4) & 31;
  int ks   = (t >> 9) & 3;
  int nt   = t >> 11;
  int col = nt * 16 + (lane & 15);
  int k   = ks * 32 + ((lane >> 4) * 8) + (j & 7) + ((j >> 3) * 16);
  float v = W[k * DIM + col];
  unsigned short h = f2bf(v);
  whi[t] = h;
  wlo[t] = f2bf(v - bf2f(h));
}

// ---------------- GEMM: H[N,128] = A[N,128] @ W[128,128], bf16x3 ----------------
__global__ void __launch_bounds__(128)
k_gemm_bf16x3(const unsigned short* __restrict__ Ahi,
              const unsigned short* __restrict__ Alo,
              const unsigned short* __restrict__ Whi,
              const unsigned short* __restrict__ Wlo,
              float* __restrict__ H, int nrows) {
  int wave = (int)((blockIdx.x * (unsigned)blockDim.x + threadIdx.x) >> 5);
  int lane = threadIdx.x & 31;
  int row0 = wave * 16;
  if (row0 >= nrows) return;                // wave-uniform
  int m = row0 + (lane & 15);
  if (m >= nrows) m = nrows - 1;            // clamp loads (tail-safe)
  int koff = (lane >> 4) * 8;
  const unsigned short* arh = Ahi + (size_t)m * DIM;
  const unsigned short* arl = Alo + (size_t)m * DIM;

  v16bf ah[KSTEPS], al[KSTEPS];
#pragma unroll
  for (int ks = 0; ks < KSTEPS; ++ks) {
    int base = ks * 32 + koff;
    v8bf h0 = *(const v8bf*)(arh + base);
    v8bf h1 = *(const v8bf*)(arh + base + 16);
    v8bf l0 = *(const v8bf*)(arl + base);
    v8bf l1 = *(const v8bf*)(arl + base + 16);
#pragma unroll
    for (int i = 0; i < 8; ++i) {
      ah[ks][i] = h0[i]; ah[ks][8 + i] = h1[i];
      al[ks][i] = l0[i]; al[ks][8 + i] = l1[i];
    }
  }

  int colw = lane & 15;
  int r0   = (lane >> 4) * 8;
#pragma unroll
  for (int nt = 0; nt < NTILES; ++nt) {
    v8f c = {};
#pragma unroll
    for (int ks = 0; ks < KSTEPS; ++ks) {
      size_t boff = (size_t)(((nt * KSTEPS + ks) * 32 + lane) * 16);
      v16bf bh = *(const v16bf*)(Whi + boff);
      v16bf bl = *(const v16bf*)(Wlo + boff);
      c = __builtin_amdgcn_wmma_f32_16x16x32_bf16(false, ah[ks], false, bh, (short)0, c, false, false);
      c = __builtin_amdgcn_wmma_f32_16x16x32_bf16(false, al[ks], false, bh, (short)0, c, false, false);
      c = __builtin_amdgcn_wmma_f32_16x16x32_bf16(false, ah[ks], false, bl, (short)0, c, false, false);
    }
#pragma unroll
    for (int r = 0; r < 8; ++r) {
      int row = row0 + r0 + r;
      if (row < nrows) H[(size_t)row * DIM + nt * 16 + colw] = c[r];
    }
  }
}

// ---------------- aggregation ----------------
__global__ void k_selfloop(const float* __restrict__ H, const float* __restrict__ dinv,
                           float* __restrict__ agg, long long total) {
  long long i = (long long)blockIdx.x * blockDim.x + threadIdx.x;
  if (i < total) {
    float di = dinv[i >> 7];
    agg[i] = H[i] * di * di;
  }
}

__global__ void k_edge_agg(const int* __restrict__ src, const int* __restrict__ dst,
                           const float* __restrict__ H, const float* __restrict__ dinv,
                           float* __restrict__ agg, int E) {
  long long t = (long long)blockIdx.x * blockDim.x + threadIdx.x;
  int e    = (int)(t >> 5);
  int lane = (int)(t & 31);
  if (e >= E) return;
  int s = src[e], d = dst[e];
  float norm = dinv[s] * dinv[d];
  float4 hv = *(const float4*)(H + (size_t)s * DIM + lane * 4);
  float* outp = agg + (size_t)d * DIM + lane * 4;
  fadd_dev(outp + 0, hv.x * norm);
  fadd_dev(outp + 1, hv.y * norm);
  fadd_dev(outp + 2, hv.z * norm);
  fadd_dev(outp + 3, hv.w * norm);
}

__global__ void k_relu_bias_split(const float* __restrict__ agg, const float* __restrict__ b,
                                  unsigned short* __restrict__ zhi,
                                  unsigned short* __restrict__ zlo, long long total) {
  long long i = (long long)blockIdx.x * blockDim.x + threadIdx.x;
  if (i < total) {
    float v = fmaxf(agg[i] + b[i & (DIM - 1)], 0.0f);
    unsigned short h = f2bf(v);
    zhi[i] = h;
    zlo[i] = f2bf(v - bf2f(h));
  }
}

__global__ void k_relu_bias_f32(float* __restrict__ agg, const float* __restrict__ b,
                                long long total) {
  long long i = (long long)blockIdx.x * blockDim.x + threadIdx.x;
  if (i < total) agg[i] = fmaxf(agg[i] + b[i & (DIM - 1)], 0.0f);
}

// ---------------- decode: out[e] = dot(Z[s], Z[d]) ----------------
__global__ void k_decode(const int* __restrict__ s, const int* __restrict__ d,
                         const float* __restrict__ Z, float* __restrict__ out, int EL) {
  long long t = (long long)blockIdx.x * blockDim.x + threadIdx.x;
  int e    = (int)(t >> 5);
  int lane = (int)(t & 31);
  if (e >= EL) return;
  float4 a  = *(const float4*)(Z + (size_t)s[e] * DIM + lane * 4);
  float4 bb = *(const float4*)(Z + (size_t)d[e] * DIM + lane * 4);
  float p = a.x * bb.x + a.y * bb.y + a.z * bb.z + a.w * bb.w;
#pragma unroll
  for (int off = 16; off > 0; off >>= 1) p += __shfl_xor(p, off, 32);
  if (lane == 0) out[e] = p;
}

// ---------------- host orchestration ----------------
extern "C" void kernel_launch(void* const* d_in, const int* in_sizes, int n_in,
                              void* d_out, int out_size, void* d_ws, size_t ws_size,
                              hipStream_t stream) {
  const float* x   = (const float*)d_in[0];
  const int*   ei  = (const int*)d_in[1];
  const int*   eli = (const int*)d_in[2];
  const float* W1  = (const float*)d_in[3];
  const float* b1  = (const float*)d_in[4];
  const float* W2  = (const float*)d_in[5];
  const float* b2  = (const float*)d_in[6];
  float* out = (float*)d_out;

  const int N  = in_sizes[0] / DIM;
  const int E  = in_sizes[1] / 2;
  const int EL = in_sizes[2] / 2;
  const int* src = ei;
  const int* dst = ei + E;
  const int* ls  = eli;
  const int* ld  = eli + EL;

  const long long NB = (long long)N * DIM;

  // workspace carve-out (256B aligned)
  char* cur = (char*)d_ws;
  auto carve = [&](size_t bytes) -> char* {
    char* p = cur;
    cur += (bytes + 255) & ~(size_t)255;
    return p;
  };
  float*          dinv = (float*)carve((size_t)N * 4);
  unsigned short* xh   = (unsigned short*)carve((size_t)NB * 2);
  unsigned short* xl   = (unsigned short*)carve((size_t)NB * 2);
  float*          h    = (float*)carve((size_t)NB * 4);
  float*          agg  = (float*)carve((size_t)NB * 4);
  unsigned short* w1h  = (unsigned short*)carve((size_t)DIM * DIM * 2);
  unsigned short* w1l  = (unsigned short*)carve((size_t)DIM * DIM * 2);
  unsigned short* w2h  = (unsigned short*)carve((size_t)DIM * DIM * 2);
  unsigned short* w2l  = (unsigned short*)carve((size_t)DIM * DIM * 2);

  const int B = 256;
  auto blocks = [&](long long n) { return (int)((n + B - 1) / B); };

  // normalization coefficients
  k_deg_init<<<blocks(N), B, 0, stream>>>((unsigned int*)dinv, N);
  k_deg_count<<<blocks(E), B, 0, stream>>>(dst, (unsigned int*)dinv, E);
  k_dinv<<<blocks(N), B, 0, stream>>>(dinv, N);

  // pack weights into WMMA B-fragment order (hi/lo split)
  const int WPN = NTILES * KSTEPS * 32 * 16;
  k_pack_w<<<blocks(WPN), B, 0, stream>>>(W1, w1h, w1l);
  k_pack_w<<<blocks(WPN), B, 0, stream>>>(W2, w2h, w2l);

  const int waves = (N + 15) / 16;
  const int gblk  = (waves * 32 + 127) / 128;

  // ---- layer 1 ----
  k_cvt_split<<<blocks(NB), B, 0, stream>>>(x, xh, xl, NB);
  k_gemm_bf16x3<<<gblk, 128, 0, stream>>>(xh, xl, w1h, w1l, h, N);
  k_selfloop<<<blocks(NB), B, 0, stream>>>(h, dinv, agg, NB);
  k_edge_agg<<<blocks((long long)E * 32), B, 0, stream>>>(src, dst, h, dinv, agg, E);
  k_relu_bias_split<<<blocks(NB), B, 0, stream>>>(agg, b1, xh, xl, NB);

  // ---- layer 2 ----
  k_gemm_bf16x3<<<gblk, 128, 0, stream>>>(xh, xl, w2h, w2l, h, N);
  k_selfloop<<<blocks(NB), B, 0, stream>>>(h, dinv, agg, NB);
  k_edge_agg<<<blocks((long long)E * 32), B, 0, stream>>>(src, dst, h, dinv, agg, E);
  k_relu_bias_f32<<<blocks(NB), B, 0, stream>>>(agg, b2, NB);

  // ---- decode ----
  k_decode<<<blocks((long long)EL * 32), B, 0, stream>>>(ls, ld, agg, out, EL);
}
